// SACCritic_67388036874466
// MI455X (gfx1250) — matrix operand order repaired and checked by reading, not compiled, not run on hardware
//
#include <hip/hip_runtime.h>
#include <hip/hip_bf16.h>

#define BB 256
#define SS 256
#define DIN 64
#define HH 512
#define LL 2
#define NHH 8
#define AACT 3
#define IDIMN 13
#define IENC 64
#define TT1 512
#define TT2 256
#define H4 2048
#define KCAT 608

typedef __bf16 bf16_t;
typedef __attribute__((ext_vector_type(16))) __bf16 v16bf;
typedef __attribute__((ext_vector_type(8))) float v8f;

// Assemble a 16-element bf16 fragment from two contiguous 16-byte chunks.
__device__ __forceinline__ v16bf frag_ld(const bf16_t* lo, const bf16_t* hi) {
  union { uint4 u[2]; v16bf v; } t;
  t.u[0] = *reinterpret_cast<const uint4*>(lo);
  t.u[1] = *reinterpret_cast<const uint4*>(hi);
  return t.v;
}

__device__ __forceinline__ v8f wmma_bf16(v16bf a, v16bf b, v8f c) {
  return __builtin_amdgcn_wmma_f32_16x16x32_bf16(false, a, false, b, (short)0, c, false, false);
}

// ---------------- weight convert: fp32 (Ksrc x N, ldsrc) -> bf16 transposed (N x Ktot), zero pad
__global__ void wconv_kernel(const float* __restrict__ src, bf16_t* __restrict__ dst,
                             int N, int Ksrc, int Koff, int Kfill, int Ktot, int ldsrc) {
  long i = (long)blockIdx.x * blockDim.x + threadIdx.x;
  long tot = (long)N * Kfill;
  if (i >= tot) return;
  int n = (int)(i / Kfill);
  int k = (int)(i % Kfill);
  float v = (k < Ksrc) ? src[(long)k * ldsrc + n] : 0.0f;
  dst[(long)n * Ktot + (Koff + k)] = (bf16_t)v;
}

__global__ void cast_f2b_kernel(const float* __restrict__ s, bf16_t* __restrict__ d, long n) {
  long i = (long)blockIdx.x * blockDim.x + threadIdx.x;
  if (i < n) d[i] = (bf16_t)s[i];
}

__global__ void infopad_kernel(const float* __restrict__ info, bf16_t* __restrict__ dst) {
  int i = blockIdx.x * blockDim.x + threadIdx.x;
  if (i >= BB * 32) return;
  int b = i >> 5, j = i & 31;
  dst[i] = (bf16_t)((j < IDIMN) ? info[b * IDIMN + j] : 0.0f);
}

// ---------------- generic bf16 WMMA GEMM: C(MxN) = A(MxK,lda) @ Bt(NxK)^T + bias
// 64x64 tile per WG, 8 waves (4Mx2N), each wave 16x32 (two 16x16x32 WMMA per K-step)
template <int OUT_BF16>
__global__ __launch_bounds__(256) void gemm_kernel(
    const bf16_t* __restrict__ Am, long lda,
    const bf16_t* __restrict__ Bt, int ldbt,
    const float* __restrict__ bias,
    void* __restrict__ Cm, int ldc, int K) {
  __shared__ bf16_t As[64 * 40];
  __shared__ bf16_t Bs[64 * 40];
  const int m0 = blockIdx.x * 64;
  const int n0 = blockIdx.y * 64;
  const int tid = threadIdx.x;
  const int w = tid >> 5, l = tid & 31;
  const int wm = (w >> 1) * 16;
  const int wn = (w & 1) * 32;
  const int lrow = tid >> 2, lcol = (tid & 3) * 8;
  const int l15 = l & 15;
  const int ka = (l < 16) ? 0 : 8;    // A: K {0..7,16..23} vs {8..15,24..31}
  const int kb2 = (l < 16) ? 0 : 8;   // B: K {0..7,16..23} vs {8..15,24..31}
  v8f acc0 = {};
  v8f acc1 = {};
  for (int kk = 0; kk < K; kk += 32) {
    uint4 av = *reinterpret_cast<const uint4*>(Am + (long)(m0 + lrow) * lda + kk + lcol);
    uint4 bv = *reinterpret_cast<const uint4*>(Bt + (long)(n0 + lrow) * ldbt + kk + lcol);
    __syncthreads();
    *reinterpret_cast<uint4*>(&As[lrow * 40 + lcol]) = av;
    *reinterpret_cast<uint4*>(&Bs[lrow * 40 + lcol]) = bv;
    __syncthreads();
    v16bf af  = frag_ld(&As[(wm + l15) * 40 + ka], &As[(wm + l15) * 40 + ka + 16]);
    v16bf bf0 = frag_ld(&Bs[(wn + l15) * 40 + kb2], &Bs[(wn + l15) * 40 + kb2 + 16]);
    v16bf bf1 = frag_ld(&Bs[(wn + 16 + l15) * 40 + kb2], &Bs[(wn + 16 + l15) * 40 + kb2 + 16]);
    acc0 = wmma_bf16(af, bf0, acc0);
    acc1 = wmma_bf16(af, bf1, acc1);
  }
  const int rbase = m0 + wm + ((l >> 4) << 3);
  const int c0 = n0 + wn + l15;
  const float b0v = bias ? bias[c0] : 0.0f;
  const float b1v = bias ? bias[c0 + 16] : 0.0f;
#pragma unroll
  for (int r = 0; r < 8; ++r) {
    long o0 = (long)(rbase + r) * ldc + c0;
    float v0 = acc0[r] + b0v;
    float v1 = acc1[r] + b1v;
    if (OUT_BF16) {
      ((bf16_t*)Cm)[o0] = (bf16_t)v0;
      ((bf16_t*)Cm)[o0 + 16] = (bf16_t)v1;
    } else {
      ((float*)Cm)[o0] = v0;
      ((float*)Cm)[o0 + 16] = v1;
    }
  }
}

// ---------------- fused sLSTM step: gates GEMM + activation + state update
// Each WG: 64 batch rows x 32 gate-columns, for ALL FOUR gates (cols {j, j+512, j+1024, j+1536}).
// gates = [xln_t | h](256x1024) @ WcatT^T + b, then z/i/f/o -> c,n,h update + residual into cur.
// 8 waves = 2(M) x 4(gate); per wave 32x32 output = 4 WMMAs per K-step.
// After the K-loop the A/B staging LDS is reused as a 64x128 fp32 gate tile for the
// cross-wave exchange needed by the pointwise phase.
__global__ __launch_bounds__(256) void slstm_step_kernel(
    const bf16_t* __restrict__ xln, bf16_t* __restrict__ hbf,
    const bf16_t* __restrict__ WcatT, const float* __restrict__ bias,
    float* __restrict__ h, float* __restrict__ c, float* __restrict__ n,
    float* __restrict__ cur, int t) {
  __shared__ char smem[64 * 128 * 4];  // 32 KB; phase1: As(5120B)+Bs(10240B); phase2: fp32 gate tile
  bf16_t* As = (bf16_t*)smem;                    // 64 x 40
  bf16_t* Bs = (bf16_t*)(smem + 64 * 40 * 2);    // 128 x 40 (4 gates x 32 cols)
  float* gt = (float*)smem;                      // 64 x 128
  const int m0 = blockIdx.x * 64;   // batch tile
  const int j0 = blockIdx.y * 32;   // gate-local column tile (0..511)
  const int tid = threadIdx.x;
  const int w = tid >> 5, l = tid & 31;
  const int wm = (w >> 2) * 32;     // 0 / 32
  const int gw = w & 3;             // gate handled by this wave
  const int arow = tid >> 2, acol = (tid & 3) * 8;
  const int brow = tid >> 1, bcol = (tid & 1) * 16;
  const int bg = brow >> 5, bjj = brow & 31;     // B loader: gate, gate-local col
  const int l15 = l & 15;
  const int ka = (l < 16) ? 0 : 8;
  const int kb2 = (l < 16) ? 0 : 8;
  v8f acc00 = {}, acc01 = {}, acc10 = {}, acc11 = {};
  for (int kk = 0; kk < 1024; kk += 32) {
    const int m = m0 + arow;
    const int k = kk + acol;
    uint4 av;
    if (k < HH)
      av = *reinterpret_cast<const uint4*>(xln + ((long)m * SS + t) * HH + k);
    else
      av = *reinterpret_cast<const uint4*>(hbf + (long)m * HH + (k - HH));
    const bf16_t* bp = WcatT + (long)(bg * HH + j0 + bjj) * 1024 + kk + bcol;
    uint4 bv0 = *reinterpret_cast<const uint4*>(bp);
    uint4 bv1 = *reinterpret_cast<const uint4*>(bp + 8);
    __syncthreads();
    *reinterpret_cast<uint4*>(&As[arow * 40 + acol]) = av;
    *reinterpret_cast<uint4*>(&Bs[brow * 40 + bcol]) = bv0;
    *reinterpret_cast<uint4*>(&Bs[brow * 40 + bcol + 8]) = bv1;
    __syncthreads();
    v16bf a0 = frag_ld(&As[(wm + l15) * 40 + ka], &As[(wm + l15) * 40 + ka + 16]);
    v16bf a1 = frag_ld(&As[(wm + 16 + l15) * 40 + ka], &As[(wm + 16 + l15) * 40 + ka + 16]);
    v16bf b0 = frag_ld(&Bs[(gw * 32 + l15) * 40 + kb2], &Bs[(gw * 32 + l15) * 40 + kb2 + 16]);
    v16bf b1 = frag_ld(&Bs[(gw * 32 + 16 + l15) * 40 + kb2], &Bs[(gw * 32 + 16 + l15) * 40 + kb2 + 16]);
    acc00 = wmma_bf16(a0, b0, acc00);
    acc01 = wmma_bf16(a0, b1, acc01);
    acc10 = wmma_bf16(a1, b0, acc10);
    acc11 = wmma_bf16(a1, b1, acc11);
  }
  __syncthreads();  // all fragment reads done; LDS now reused as fp32 gate tile
  const int roff = (l >> 4) << 3;
  const float bv0 = bias[gw * HH + j0 + l15];
  const float bv1 = bias[gw * HH + j0 + 16 + l15];
#pragma unroll
  for (int r = 0; r < 8; ++r) {
    int row0 = wm + r + roff;
    int col0 = gw * 32 + l15;
    gt[row0 * 128 + col0] = acc00[r] + bv0;
    gt[row0 * 128 + col0 + 16] = acc01[r] + bv1;
    gt[(row0 + 16) * 128 + col0] = acc10[r] + bv0;
    gt[(row0 + 16) * 128 + col0 + 16] = acc11[r] + bv1;
  }
  __syncthreads();
  // pointwise sLSTM update: thread handles 1 row x 8 gate-local columns
  const int prow = tid >> 2;
  const int pjb = (tid & 3) * 8;
  const int b = m0 + prow;
#pragma unroll
  for (int q = 0; q < 8; ++q) {
    int jj = pjb + q;
    int j = j0 + jj;
    float z  = tanhf(gt[prow * 128 + jj]);
    float ii = expf(gt[prow * 128 + 32 + jj]);
    float ff = 1.0f / (1.0f + expf(-gt[prow * 128 + 64 + jj]));
    float oo = 1.0f / (1.0f + expf(-gt[prow * 128 + 96 + jj]));
    long si = (long)b * HH + j;
    float c2 = ff * c[si] + ii * z;
    float n2 = ff * n[si] + ii;
    float h2 = oo * c2 / n2;
    c[si] = c2; n[si] = n2; h[si] = h2;
    hbf[si] = (bf16_t)h2;
    cur[((long)b * SS + t) * HH + j] += h2;  // out = x_t + h2 (in-place)
  }
}

__global__ void init_state_kernel(float* h, float* c, float* n, bf16_t* hbf) {
  int i = blockIdx.x * blockDim.x + threadIdx.x;
  if (i >= BB * HH) return;
  h[i] = 0.f; c[i] = 0.f; n[i] = 1.f; hbf[i] = (bf16_t)0.f;
}

// ---------------- LayerNorm (+optional exact GELU); one row per block
__global__ __launch_bounds__(256) void ln_kernel(const float* __restrict__ src,
                                                 const float* __restrict__ g,
                                                 const float* __restrict__ bta,
                                                 float* __restrict__ dstF,
                                                 bf16_t* __restrict__ dstB,
                                                 int N, int gelu) {
  long row = blockIdx.x;
  const float* x = src + row * N;
  __shared__ float red[256];
  float vals[4];
  int tid = threadIdx.x;
  float s = 0.f, ss = 0.f;
  int cnt = 0;
  for (int cIdx = tid; cIdx < N; cIdx += 256) {
    float v = x[cIdx];
    vals[cnt++] = v;
    s += v; ss += v * v;
  }
  red[tid] = s; __syncthreads();
  for (int o = 128; o > 0; o >>= 1) { if (tid < o) red[tid] += red[tid + o]; __syncthreads(); }
  float mean = red[0] / N;
  __syncthreads();
  red[tid] = ss; __syncthreads();
  for (int o = 128; o > 0; o >>= 1) { if (tid < o) red[tid] += red[tid + o]; __syncthreads(); }
  float var = red[0] / N - mean * mean;
  float rs = rsqrtf(var + 1e-5f);
  cnt = 0;
  for (int cIdx = tid; cIdx < N; cIdx += 256) {
    float y = (vals[cnt++] - mean) * rs * g[cIdx] + bta[cIdx];
    if (gelu) y = 0.5f * y * (1.0f + erff(y * 0.70710678118654752f));
    if (dstF) dstF[row * N + cIdx] = y;
    if (dstB) dstB[row * N + cIdx] = (bf16_t)y;
  }
}

// ---------------- attention with only the last query: one block per (b,head)
__global__ __launch_bounds__(256) void attn_kernel(const float* __restrict__ qlast,
                                                   const bf16_t* __restrict__ Kb,
                                                   const bf16_t* __restrict__ Vb,
                                                   bf16_t* __restrict__ ao) {
  int b = blockIdx.x >> 3;
  int h = blockIdx.x & 7;
  int s = threadIdx.x;
  __shared__ float sc[256];
  __shared__ float red[256];
  const float* q = qlast + b * HH + h * 64;
  const bf16_t* kp = Kb + ((long)b * SS + s) * HH + h * 64;
  float dot = 0.f;
#pragma unroll 8
  for (int d = 0; d < 64; ++d) dot += q[d] * (float)kp[d];
  dot *= 0.125f;  // dh^-0.5, dh=64
  sc[s] = dot; red[s] = dot; __syncthreads();
  for (int o = 128; o > 0; o >>= 1) { if (s < o) red[s] = fmaxf(red[s], red[s + o]); __syncthreads(); }
  float mx = red[0];
  __syncthreads();
  float e = expf(sc[s] - mx);
  sc[s] = e; red[s] = e; __syncthreads();
  for (int o = 128; o > 0; o >>= 1) { if (s < o) red[s] += red[s + o]; __syncthreads(); }
  float inv = 1.0f / red[0];
  __syncthreads();
  if (s < 64) {
    const bf16_t* vp = Vb + (long)b * SS * HH + h * 64 + s;
    float acc = 0.f;
    for (int t2 = 0; t2 < SS; ++t2) acc += sc[t2] * (float)vp[(long)t2 * HH];
    ao[b * HH + h * 64 + s] = (bf16_t)(acc * inv);
  }
}

__global__ void cat_kernel(const float* __restrict__ context, const bf16_t* __restrict__ ie,
                           const float* __restrict__ action, bf16_t* __restrict__ cat) {
  int i = blockIdx.x * blockDim.x + threadIdx.x;
  if (i >= BB * KCAT) return;
  int b = i / KCAT, j = i % KCAT;
  if (j < HH) { cat[i] = (bf16_t)context[b * HH + j]; return; }
  if (j < HH + IENC) { cat[i] = ie[b * IENC + (j - HH)]; return; }
  if (j < HH + IENC + AACT) { cat[i] = (bf16_t)action[b * AACT + (j - HH - IENC)]; return; }
  cat[i] = (bf16_t)0.f;
}

__global__ void heads_kernel(const float* __restrict__ f2, const float* __restrict__ w1,
                             const float* __restrict__ b1, const float* __restrict__ w2,
                             const float* __restrict__ b2, float* __restrict__ out) {
  int b = blockIdx.x * blockDim.x + threadIdx.x;
  if (b >= BB) return;
  float s1 = b1[0], s2 = b2[0];
  for (int j = 0; j < TT2; ++j) {
    float f = f2[b * TT2 + j];
    s1 += f * w1[j];
    s2 += f * w2[j];
  }
  out[b] = s1;
  out[BB + b] = s2;
}

__global__ void copy_state_kernel(const float* __restrict__ h, const float* __restrict__ c,
                                  const float* __restrict__ n, float* __restrict__ out, int l) {
  int i = blockIdx.x * blockDim.x + threadIdx.x;
  if (i >= BB * HH) return;
  const long base = 2 * BB;         // q1,q2
  const long bh = (long)BB * HH;
  out[base + (long)l * bh + i] = h[i];
  out[base + LL * bh + (long)l * bh + i] = c[i];
  out[base + 2 * LL * bh + (long)l * bh + i] = n[i];
}

extern "C" void kernel_launch(void* const* d_in, const int* in_sizes, int n_in,
                              void* d_out, int out_size, void* d_ws, size_t ws_size,
                              hipStream_t stream) {
  (void)in_sizes; (void)n_in; (void)out_size; (void)ws_size;
  const float* x       = (const float*)d_in[0];
  const float* action  = (const float*)d_in[1];
  const float* info    = (const float*)d_in[2];
  const float* in_W    = (const float*)d_in[3];
  const float* in_b    = (const float*)d_in[4];
  const float* in_g    = (const float*)d_in[5];
  const float* in_beta = (const float*)d_in[6];
  const float* cell_W  = (const float*)d_in[7];
  const float* cell_R  = (const float*)d_in[8];
  const float* cell_b  = (const float*)d_in[9];
  const float* lnl_g   = (const float*)d_in[10];
  const float* lnl_b   = (const float*)d_in[11];
  const float* Wq = (const float*)d_in[12];
  const float* Wk = (const float*)d_in[13];
  const float* Wv = (const float*)d_in[14];
  const float* Wo = (const float*)d_in[15];
  const float* bq = (const float*)d_in[16];
  const float* bk = (const float*)d_in[17];
  const float* bv = (const float*)d_in[18];
  const float* bo = (const float*)d_in[19];
  const float* iW1 = (const float*)d_in[20];
  const float* ib1 = (const float*)d_in[21];
  const float* ig1 = (const float*)d_in[22];
  const float* ibt1 = (const float*)d_in[23];
  const float* iW2 = (const float*)d_in[24];
  const float* ib2 = (const float*)d_in[25];
  const float* ig2 = (const float*)d_in[26];
  const float* ibt2 = (const float*)d_in[27];
  const float* sW1 = (const float*)d_in[28];
  const float* sb1 = (const float*)d_in[29];
  const float* sg1 = (const float*)d_in[30];
  const float* sbt1 = (const float*)d_in[31];
  const float* sW2 = (const float*)d_in[32];
  const float* sb2 = (const float*)d_in[33];
  const float* sg2 = (const float*)d_in[34];
  const float* sbt2 = (const float*)d_in[35];
  const float* q1W = (const float*)d_in[36];
  const float* q1b = (const float*)d_in[37];
  const float* q2W = (const float*)d_in[38];
  const float* q2b = (const float*)d_in[39];
  float* outp = (float*)d_out;

  char* ws = (char*)d_ws;
  size_t off = 0;
  auto alloc = [&](size_t bytes) -> char* {
    char* p = ws + off;
    off = (off + bytes + 255) & ~(size_t)255;
    return p;
  };
  float*  cur    = (float*) alloc((size_t)BB * SS * HH * 4);
  bf16_t* xln    = (bf16_t*)alloc((size_t)BB * SS * HH * 2);  // reused as attn input bf16
  bf16_t* kbuf   = (bf16_t*)alloc((size_t)BB * SS * HH * 2);
  bf16_t* vbuf   = (bf16_t*)alloc((size_t)BB * SS * HH * 2);
  float*  hstate = (float*) alloc((size_t)BB * HH * 4);
  float*  cstate = (float*) alloc((size_t)BB * HH * 4);
  float*  nstate = (float*) alloc((size_t)BB * HH * 4);
  bf16_t* hbf    = (bf16_t*)alloc((size_t)BB * HH * 2);
  bf16_t* xbf    = (bf16_t*)alloc((size_t)BB * SS * DIN * 2);
  float*  qlast  = (float*) alloc((size_t)BB * HH * 4);
  bf16_t* aobf   = (bf16_t*)alloc((size_t)BB * HH * 2);
  float*  context= (float*) alloc((size_t)BB * HH * 4);
  bf16_t* infobf = (bf16_t*)alloc((size_t)BB * 32 * 2);
  float*  ie1    = (float*) alloc((size_t)BB * IENC * 4);
  bf16_t* ie1b   = (bf16_t*)alloc((size_t)BB * IENC * 2);
  float*  ie2    = (float*) alloc((size_t)BB * IENC * 4);
  bf16_t* ie2b   = (bf16_t*)alloc((size_t)BB * IENC * 2);
  bf16_t* catbf  = (bf16_t*)alloc((size_t)BB * KCAT * 2);
  float*  f1     = (float*) alloc((size_t)BB * TT1 * 4);
  bf16_t* f1b    = (bf16_t*)alloc((size_t)BB * TT1 * 2);
  float*  f2     = (float*) alloc((size_t)BB * TT2 * 4);
  float*  f2g    = (float*) alloc((size_t)BB * TT2 * 4);
  bf16_t* inWt   = (bf16_t*)alloc((size_t)HH * DIN * 2);
  bf16_t* WcatT  = (bf16_t*)alloc((size_t)LL * H4 * 1024 * 2);
  bf16_t* WqT    = (bf16_t*)alloc((size_t)HH * HH * 2);
  bf16_t* WkT    = (bf16_t*)alloc((size_t)HH * HH * 2);
  bf16_t* WvT    = (bf16_t*)alloc((size_t)HH * HH * 2);
  bf16_t* WoT    = (bf16_t*)alloc((size_t)HH * HH * 2);
  bf16_t* iW1t   = (bf16_t*)alloc((size_t)IENC * 32 * 2);
  bf16_t* iW2t   = (bf16_t*)alloc((size_t)IENC * IENC * 2);
  bf16_t* sW1t   = (bf16_t*)alloc((size_t)TT1 * KCAT * 2);
  bf16_t* sW2t   = (bf16_t*)alloc((size_t)TT2 * TT1 * 2);

  auto cdiv = [](long a, long b) { return (int)((a + b - 1) / b); };

  // --- weight conversion (fp32 -> bf16, transposed to N-major) ---
  wconv_kernel<<<cdiv((long)HH * DIN, 256), 256, 0, stream>>>(in_W, inWt, HH, DIN, 0, DIN, DIN, HH);
  for (int l2 = 0; l2 < LL; ++l2) {
    bf16_t* dst = WcatT + (size_t)l2 * H4 * 1024;
    wconv_kernel<<<cdiv((long)H4 * HH, 256), 256, 0, stream>>>(cell_W + (size_t)l2 * HH * H4, dst, H4, HH, 0, HH, 1024, H4);
    wconv_kernel<<<cdiv((long)H4 * HH, 256), 256, 0, stream>>>(cell_R + (size_t)l2 * HH * H4, dst, H4, HH, HH, HH, 1024, H4);
  }
  wconv_kernel<<<cdiv((long)HH * HH, 256), 256, 0, stream>>>(Wq, WqT, HH, HH, 0, HH, HH, HH);
  wconv_kernel<<<cdiv((long)HH * HH, 256), 256, 0, stream>>>(Wk, WkT, HH, HH, 0, HH, HH, HH);
  wconv_kernel<<<cdiv((long)HH * HH, 256), 256, 0, stream>>>(Wv, WvT, HH, HH, 0, HH, HH, HH);
  wconv_kernel<<<cdiv((long)HH * HH, 256), 256, 0, stream>>>(Wo, WoT, HH, HH, 0, HH, HH, HH);
  wconv_kernel<<<cdiv((long)IENC * 32, 256), 256, 0, stream>>>(iW1, iW1t, IENC, IDIMN, 0, 32, 32, IENC);
  wconv_kernel<<<cdiv((long)IENC * IENC, 256), 256, 0, stream>>>(iW2, iW2t, IENC, IENC, 0, IENC, IENC, IENC);
  wconv_kernel<<<cdiv((long)TT1 * KCAT, 256), 256, 0, stream>>>(sW1, sW1t, TT1, HH + IENC + AACT, 0, KCAT, KCAT, TT1);
  wconv_kernel<<<cdiv((long)TT2 * TT1, 256), 256, 0, stream>>>(sW2, sW2t, TT2, TT1, 0, TT1, TT1, TT2);

  // --- input projection + LN ---
  cast_f2b_kernel<<<cdiv((long)BB * SS * DIN, 256), 256, 0, stream>>>(x, xbf, (long)BB * SS * DIN);
  gemm_kernel<0><<<dim3((BB * SS) / 64, HH / 64), 256, 0, stream>>>(xbf, DIN, inWt, DIN, in_b, cur, HH, DIN);
  ln_kernel<<<BB * SS, 256, 0, stream>>>(cur, in_g, in_beta, cur, nullptr, HH, 0);

  // --- sLSTM layers (sequential scan, fused GEMM + gate/state update per step) ---
  for (int l2 = 0; l2 < LL; ++l2) {
    ln_kernel<<<BB * SS, 256, 0, stream>>>(cur, lnl_g + l2 * HH, lnl_b + l2 * HH, nullptr, xln, HH, 0);
    init_state_kernel<<<cdiv(BB * HH, 256), 256, 0, stream>>>(hstate, cstate, nstate, hbf);
    const bf16_t* Wc = WcatT + (size_t)l2 * H4 * 1024;
    const float* gb = cell_b + (size_t)l2 * H4;
    for (int t = 0; t < SS; ++t) {
      slstm_step_kernel<<<dim3(BB / 64, HH / 32), 256, 0, stream>>>(
          xln, hbf, Wc, gb, hstate, cstate, nstate, cur, t);
    }
    copy_state_kernel<<<cdiv(BB * HH, 256), 256, 0, stream>>>(hstate, cstate, nstate, outp, l2);
  }

  // --- attention (only last-position query needed) ---
  cast_f2b_kernel<<<cdiv((long)BB * SS * HH, 256), 256, 0, stream>>>(cur, xln, (long)BB * SS * HH);
  gemm_kernel<0><<<dim3(BB / 64, HH / 64), 256, 0, stream>>>(xln + (size_t)(SS - 1) * HH, (long)SS * HH, WqT, HH, bq, qlast, HH, HH);
  gemm_kernel<1><<<dim3((BB * SS) / 64, HH / 64), 256, 0, stream>>>(xln, HH, WkT, HH, bk, kbuf, HH, HH);
  gemm_kernel<1><<<dim3((BB * SS) / 64, HH / 64), 256, 0, stream>>>(xln, HH, WvT, HH, bv, vbuf, HH, HH);
  attn_kernel<<<BB * NHH, 256, 0, stream>>>(qlast, kbuf, vbuf, aobf);
  gemm_kernel<0><<<dim3(BB / 64, HH / 64), 256, 0, stream>>>(aobf, HH, WoT, HH, bo, context, HH, HH);

  // --- info encoder ---
  infopad_kernel<<<cdiv(BB * 32, 256), 256, 0, stream>>>(info, infobf);
  gemm_kernel<0><<<dim3(BB / 64, IENC / 64), 256, 0, stream>>>(infobf, 32, iW1t, 32, ib1, ie1, IENC, 32);
  ln_kernel<<<BB, 256, 0, stream>>>(ie1, ig1, ibt1, nullptr, ie1b, IENC, 1);
  gemm_kernel<0><<<dim3(BB / 64, IENC / 64), 256, 0, stream>>>(ie1b, IENC, iW2t, IENC, ib2, ie2, IENC, IENC);
  ln_kernel<<<BB, 256, 0, stream>>>(ie2, ig2, ibt2, nullptr, ie2b, IENC, 1);

  // --- trunk MLP + heads ---
  cat_kernel<<<cdiv(BB * KCAT, 256), 256, 0, stream>>>(context, ie2b, action, catbf);
  gemm_kernel<0><<<dim3(BB / 64, TT1 / 64), 256, 0, stream>>>(catbf, KCAT, sW1t, KCAT, sb1, f1, TT1, KCAT);
  ln_kernel<<<BB, 256, 0, stream>>>(f1, sg1, sbt1, nullptr, f1b, TT1, 1);
  gemm_kernel<0><<<dim3(BB / 64, TT2 / 64), 256, 0, stream>>>(f1b, TT1, sW2t, TT1, sb2, f2, TT2, TT1);
  ln_kernel<<<BB, 256, 0, stream>>>(f2, sg2, sbt2, f2g, nullptr, TT2, 1);
  heads_kernel<<<cdiv(BB, 256), 256, 0, stream>>>(f2g, q1W, q1b, q2W, q2b, outp);
}
